// DKFN_73770358276543
// MI455X (gfx1250) — matrix operand order repaired and checked by reading, not compiled, not run on hardware
//
#include <hip/hip_runtime.h>
#include <hip/hip_bf16.h>
#include <stdint.h>

typedef __bf16 bf16;
typedef __attribute__((ext_vector_type(16))) __bf16 v16bf;
typedef __attribute__((ext_vector_type(8)))  __bf16 v8bf;
typedef __attribute__((ext_vector_type(8)))  float  v8f;

#define NND   1024        // N
#define BCR   128         // B*C rows
#define LSTEP 24
#define GATE_N 4096       // 4*N
#define GC_N   3072       // K*N
#define RIN_K  2048       // 2*N

__device__ __forceinline__ float sigm(float x) { return 1.0f / (1.0f + __expf(-x)); }

// ------------------------------------------------------------------
// Precompute: column sums of adj (tmp), clamped to 1e-5 on zero
// ------------------------------------------------------------------
__global__ void colsum_kernel(const float* __restrict__ adj, float* __restrict__ tmp) {
    int j = blockIdx.x * blockDim.x + threadIdx.x;
    if (j >= NND) return;
    float s = 0.f;
    for (int i = 0; i < NND; ++i) s += adj[i * NND + j];
    tmp[j] = (s == 0.f) ? 1e-5f : s;
}

// a_norm[i][j] = tmp[i] * adj[i][j]
__global__ void anorm_kernel(const float* __restrict__ adj, const float* __restrict__ tmp,
                             float* __restrict__ anorm) {
    int idx = blockIdx.x * blockDim.x + threadIdx.x;  // NND*NND
    int i = idx >> 10;
    anorm[idx] = tmp[i] * adj[idx];
}

// A1 = min(a_norm, 1)
__global__ void clamp1_kernel(const float* __restrict__ src, float* __restrict__ dst) {
    int idx = blockIdx.x * blockDim.x + threadIdx.x;
    dst[idx] = fminf(src[idx], 1.0f);
}

// C = min(A @ B, 1), square n x n, fp32 (one-time precompute; tiny FLOPs)
#define TS 16
__global__ __launch_bounds__(256) void matmul_clamp_kernel(const float* __restrict__ A,
                                                           const float* __restrict__ B,
                                                           float* __restrict__ C, int n) {
    __shared__ float As[TS][TS];
    __shared__ float Bs[TS][TS + 1];
    int tx = threadIdx.x, ty = threadIdx.y;
    int row = blockIdx.y * TS + ty, col = blockIdx.x * TS + tx;
    float acc = 0.f;
    for (int k0 = 0; k0 < n; k0 += TS) {
        As[ty][tx] = A[(size_t)row * n + k0 + tx];
        Bs[ty][tx] = B[(size_t)(k0 + ty) * n + col];
        __syncthreads();
#pragma unroll
        for (int kk = 0; kk < TS; ++kk) acc += As[ty][kk] * Bs[kk][tx];
        __syncthreads();
    }
    C[(size_t)row * n + col] = fminf(acc, 1.0f);
}

// nbr[m] = sum_n A3[m][n] * nw[n]
__global__ void nbr_kernel(const float* __restrict__ A3, const float* __restrict__ nw,
                           float* __restrict__ nbr) {
    int m = blockIdx.x * blockDim.x + threadIdx.x;
    if (m >= NND) return;
    float s = 0.f;
    for (int n = 0; n < NND; ++n) s += A3[(size_t)m * NND + n] * nw[n];
    nbr[m] = s;
}

// ------------------------------------------------------------------
// Weight packing into fragment-ready bf16 B layout.
// Layout: tile (kt, nt) of 32(K) x 16(Nout); flat index
//   (((kt*ntiles + nt)*32 + lane)*16 + e)
// lane: col = lane&15, K = (lane>>4)*16 + e   (e in [0,16))
// ------------------------------------------------------------------
__global__ void pack_effw_kernel(const float* __restrict__ A1, const float* __restrict__ A2,
                                 const float* __restrict__ A3, const float* __restrict__ gc_w,
                                 bf16* __restrict__ dst) {
    long tid = (long)blockIdx.x * blockDim.x + threadIdx.x;
    const long total = (long)NND * GC_N;
    if (tid >= total) return;
    int  e = tid & 15;
    int  lane = (tid >> 4) & 31;
    long tileIdx = tid >> 9;
    const int ntiles = GC_N / 16;  // 192
    int nt = (int)(tileIdx % ntiles);
    int kt = (int)(tileIdx / ntiles);
    int j = nt * 16 + (lane & 15);              // output col in [0, 3N)
    int kl = ((lane >> 4) << 4) + e;            // 0..31
    int i = kt * 32 + kl;                       // input (n) index
    int kf = j >> 10, m = j & 1023;
    const float* Ak = (kf == 0) ? A1 : ((kf == 1) ? A2 : A3);
    float v = Ak[(size_t)m * NND + i] * gc_w[((size_t)kf * NND + m) * NND + i];
    dst[tid] = (bf16)v;
}

__global__ void pack_gate_kernel(const float* __restrict__ Wf, const float* __restrict__ Wi,
                                 const float* __restrict__ Wo, const float* __restrict__ Wc,
                                 bf16* __restrict__ dst) {
    long tid = (long)blockIdx.x * blockDim.x + threadIdx.x;
    const long total = (long)GATE_N * GATE_N;  // Kin=4096, Nout=4096
    if (tid >= total) return;
    int  e = tid & 15;
    int  lane = (tid >> 4) & 31;
    long tileIdx = tid >> 9;
    const int ntiles = GATE_N / 16;  // 256
    int nt = (int)(tileIdx % ntiles);
    int kt = (int)(tileIdx / ntiles);
    int j = nt * 16 + (lane & 15);
    int kl = ((lane >> 4) << 4) + e;
    int i = kt * 32 + kl;
    int sel = j >> 10, r = j & 1023;
    const float* W = (sel == 0) ? Wf : (sel == 1) ? Wi : (sel == 2) ? Wo : Wc;
    dst[tid] = (bf16)W[(size_t)r * GATE_N + i];
}

__global__ void pack_rgate_kernel(const float* __restrict__ Wf, const float* __restrict__ Wi,
                                  const float* __restrict__ Wo, const float* __restrict__ Wc,
                                  bf16* __restrict__ dst) {
    long tid = (long)blockIdx.x * blockDim.x + threadIdx.x;
    const long total = (long)RIN_K * GATE_N;  // Kin=2048, Nout=4096
    if (tid >= total) return;
    int  e = tid & 15;
    int  lane = (tid >> 4) & 31;
    long tileIdx = tid >> 9;
    const int ntiles = GATE_N / 16;  // 256
    int nt = (int)(tileIdx % ntiles);
    int kt = (int)(tileIdx / ntiles);
    int j = nt * 16 + (lane & 15);
    int kl = ((lane >> 4) << 4) + e;
    int i = kt * 32 + kl;
    int sel = j >> 10, r = j & 1023;
    const float* W = (sel == 0) ? Wf : (sel == 1) ? Wi : (sel == 2) ? Wo : Wc;
    dst[tid] = (bf16)W[(size_t)r * RIN_K + i];
}

__global__ void pack_bias_kernel(const float* bf_, const float* bi_, const float* bo_, const float* bc_,
                                 const float* rbf_, const float* rbi_, const float* rbo_, const float* rbc_,
                                 float* __restrict__ BG, float* __restrict__ RBG) {
    int j = blockIdx.x * blockDim.x + threadIdx.x;
    if (j >= GATE_N) return;
    int sel = j >> 10, r = j & 1023;
    BG[j]  = ((sel == 0) ? bf_  : (sel == 1) ? bi_  : (sel == 2) ? bo_  : bc_ )[r];
    RBG[j] = ((sel == 0) ? rbf_ : (sel == 1) ? rbi_ : (sel == 2) ? rbo_ : rbc_)[r];
}

__global__ void zero_u32_kernel(uint32_t* __restrict__ p, int n) {
    int i = blockIdx.x * blockDim.x + threadIdx.x;
    if (i < n) p[i] = 0u;
}

// ------------------------------------------------------------------
// bf16 WMMA GEMM: C(M=128 x Nn) = A(128 x Kk, row-major bf16) * Bpk (packed)
// block = 128 threads (4 waves); each wave computes a 32x64 strip
// (2 m-tiles x 4 n-tiles -> 8 WMMAs per k-tile) so every packed B tile is
// reused across 2 m-tiles, halving L2 B traffic vs 1 m-tile/wave.
// A tile (32x32 bf16, 2KB) staged per-wave through LDS with
// global_load_async_to_lds_b128 double-buffering (ASYNCcnt pipelining);
// B fragments are register double-buffered so WMMAs overlap loads.
// grid = (Nn/256, 4)
// ------------------------------------------------------------------
__global__ __launch_bounds__(128) void gemm_bf16_wmma(
    const bf16* __restrict__ A, const bf16* __restrict__ Bpk,
    const float* __restrict__ bias, float* __restrict__ Cf,
    bf16* __restrict__ Cb, int ldcb, int Kk, int Nn) {
    __shared__ bf16 smA[4][2][1024];  // [wave][buf][32 rows x 32 k], 2KB per buf
    const int lane = threadIdx.x & 31;
    const int wave = threadIdx.x >> 5;
    const int m0 = blockIdx.y * 32;
    const int n0 = blockIdx.x * 256 + wave * 64;
    const int ntiles = Nn >> 4;
    const int nt0 = n0 >> 4;

    // --- async A staging addressing ---
    // tile = 32 rows x 64 bytes; each lane moves 16B per issue, 4 issues per tile
    const uint32_t ldsA0 = (uint32_t)(uintptr_t)(&smA[wave][0][0]) + (uint32_t)(lane * 16);
    const uint32_t ldsA1 = (uint32_t)(uintptr_t)(&smA[wave][1][0]) + (uint32_t)(lane * 16);
    const bf16* Ab = A + (size_t)m0 * Kk;

    auto issueA = [&](int kt, uint32_t ldsBase) {
        // order LDS fragment reads of the buffer being overwritten before re-issue
        asm volatile("s_wait_dscnt 0x0" ::: "memory");
#pragma unroll
        for (int q = 0; q < 4; ++q) {
            int off = q * 512 + lane * 16;        // linear byte offset in 2KB tile
            int row = off >> 6;                   // 64B per row
            int el = (off & 63) >> 1;             // bf16 element in row
            const bf16* g = Ab + (size_t)row * Kk + kt * 32 + el;
            uint32_t l = ldsBase + (uint32_t)(q * 512);
            asm volatile("global_load_async_to_lds_b128 %0, %1, off"
                         :: "v"(l), "v"(g) : "memory");
        }
    };

    auto readA = [&](const bf16* smbuf, int half) -> v16bf {
        const int row = lane & 15;
        const int kb = (lane >> 4) << 4;  // 0 or 16
        const v8bf* lp = (const v8bf*)(smbuf + half * 512 + row * 32 + kb);
        v8bf lo = lp[0];
        v8bf hi = lp[1];  // +8 elements
        return __builtin_shufflevector(lo, hi, 0, 1, 2, 3, 4, 5, 6, 7,
                                       8, 9, 10, 11, 12, 13, 14, 15);
    };

    auto loadB = [&](int kt, v16bf* B) {
        const bf16* bbase = Bpk + (((size_t)kt * ntiles + nt0) * 32 + lane) * 16;
#pragma unroll
        for (int s = 0; s < 4; ++s) {
            const v8bf* bp = (const v8bf*)(bbase + (size_t)s * 512);
            v8bf lo = bp[0];
            v8bf hi = bp[1];
            B[s] = __builtin_shufflevector(lo, hi, 0, 1, 2, 3, 4, 5, 6, 7,
                                           8, 9, 10, 11, 12, 13, 14, 15);
        }
    };

    v8f acc[2][4];
#pragma unroll
    for (int mi = 0; mi < 2; ++mi)
#pragma unroll
        for (int s = 0; s < 4; ++s)
            acc[mi][s] = (v8f){0.f, 0.f, 0.f, 0.f, 0.f, 0.f, 0.f, 0.f};

    v16bf Bp[4], Bq[4];
    const int cnt = Kk >> 5;  // multiple of 2 for all call sites
    issueA(0, ldsA0);
    loadB(0, Bp);

    for (int kt = 0; kt < cnt; kt += 2) {
        // ---- half 1: compute kt (A buf0, Bp) while fetching kt+1 ----
        issueA(kt + 1, ldsA1);
        loadB(kt + 1, Bq);
        asm volatile("s_wait_asynccnt 0x4" ::: "memory");  // tile kt resident
#pragma unroll
        for (int mi = 0; mi < 2; ++mi) {
            v16bf a = readA(&smA[wave][0][0], mi);
#pragma unroll
            for (int s = 0; s < 4; ++s)
                acc[mi][s] = __builtin_amdgcn_wmma_f32_16x16x32_bf16(
                    false, a, false, Bp[s], (short)0, acc[mi][s], false, false);
        }
        // ---- half 2: compute kt+1 (A buf1, Bq) while fetching kt+2 ----
        if (kt + 2 < cnt) {
            issueA(kt + 2, ldsA0);
            loadB(kt + 2, Bp);
            asm volatile("s_wait_asynccnt 0x4" ::: "memory");  // tile kt+1 resident
        } else {
            asm volatile("s_wait_asynccnt 0x0" ::: "memory");
        }
#pragma unroll
        for (int mi = 0; mi < 2; ++mi) {
            v16bf a = readA(&smA[wave][1][0], mi);
#pragma unroll
            for (int s = 0; s < 4; ++s)
                acc[mi][s] = __builtin_amdgcn_wmma_f32_16x16x32_bf16(
                    false, a, false, Bq[s], (short)0, acc[mi][s], false, false);
        }
    }

    // C/D layout: VGPR r: lanes 0-15 -> M = r, lanes 16-31 -> M = 8 + r; col = lane&15
    const int col = lane & 15;
    const int rbase = (lane >> 4) << 3;
#pragma unroll
    for (int mi = 0; mi < 2; ++mi) {
#pragma unroll
        for (int s = 0; s < 4; ++s) {
            int ncol = n0 + s * 16 + col;
            float bv = bias ? bias[ncol] : 0.f;
#pragma unroll
            for (int r = 0; r < 8; ++r) {
                int mrow = m0 + mi * 16 + rbase + r;
                float v = acc[mi][s][r] + bv;
                Cf[(size_t)mrow * Nn + ncol] = v;
                if (Cb) Cb[(size_t)mrow * ldcb + ncol] = (bf16)v;
            }
        }
    }
}

// ------------------------------------------------------------------
// Per-step helpers
// ------------------------------------------------------------------
// cast x_t into XA (128x1024 bf16) and RA[:, 0:1024]
__global__ void cast_x_kernel(const float* __restrict__ inputs, int t,
                              bf16* __restrict__ XA, bf16* __restrict__ RA) {
    int idx = blockIdx.x * blockDim.x + threadIdx.x;  // 128*1024
    if (idx >= BCR * NND) return;
    int bc = idx >> 10, n = idx & 1023;
    float v = inputs[(size_t)idx * LSTEP + t];
    bf16 h = (bf16)v;
    XA[idx] = h;
    RA[(size_t)bc * RIN_K + n] = h;
}

// sum / sumsq accumulation per channel (row&1) with block reduction + atomics
__global__ __launch_bounds__(256) void var_acc_kernel(const float* __restrict__ src, int cols,
                                                      int stride, int off,
                                                      float* __restrict__ accum) {
    __shared__ float ssum[256];
    __shared__ float ssq[256];
    int row = blockIdx.y;                       // 0..127
    int col = blockIdx.x * 256 + threadIdx.x;   // cols multiple of 256
    int c = row & 1;
    float v = src[((size_t)row * cols + col) * stride + off];
    ssum[threadIdx.x] = v;
    ssq[threadIdx.x] = v * v;
    __syncthreads();
    for (int s = 128; s > 0; s >>= 1) {
        if (threadIdx.x < (unsigned)s) {
            ssum[threadIdx.x] += ssum[threadIdx.x + s];
            ssq[threadIdx.x] += ssq[threadIdx.x + s];
        }
        __syncthreads();
    }
    if (threadIdx.x == 0) {
        atomicAdd(&accum[c], ssum[0]);
        atomicAdd(&accum[2 + c], ssq[0]);
    }
}

// LSTM / DKFN state update + pred; also writes bf16 H, rH for next step
__global__ void lstm_step_kernel(const float* __restrict__ G, const float* __restrict__ RG,
                                 const float* __restrict__ nbr, float* __restrict__ H,
                                 float* __restrict__ Cs, float* __restrict__ rH,
                                 float* __restrict__ rCs, float* __restrict__ pred,
                                 bf16* __restrict__ GA, bf16* __restrict__ RA,
                                 const float* __restrict__ var_accum,
                                 const float* __restrict__ c_scalar) {
    int idx = blockIdx.x * blockDim.x + threadIdx.x;
    if (idx >= BCR * NND) return;
    int bc = idx >> 10, n = idx & 1023, c = bc & 1;

    const float CNT_X = 65536.f;    // B*N per channel
    const float CNT_GC = 196608.f;  // B*3N per channel
    float sx = var_accum[c], sxx = var_accum[2 + c];
    float var1 = (sxx - sx * sx / CNT_X) / (CNT_X - 1.f);
    float sg = var_accum[4 + c], sgg = var_accum[6 + c];
    float var2 = (sgg - sg * sg / CNT_GC) / (CNT_GC - 1.f);

    const float* g = G + (size_t)bc * GATE_N;
    float f  = sigm(g[n]);
    float i_ = sigm(g[NND + n]);
    float o  = sigm(g[2 * NND + n]);
    float ct = tanhf(g[3 * NND + n]);
    float cs = f * (Cs[idx] * nbr[n]) + i_ * ct;
    float h = o * tanhf(cs);

    const float* rg = RG + (size_t)bc * GATE_N;
    float rf  = sigm(rg[n]);
    float ri  = sigm(rg[NND + n]);
    float ro  = sigm(rg[2 * NND + n]);
    float rct = tanhf(rg[3 * NND + n]);
    float rcs = rf * rCs[idx] + ri * rct;
    float rh = ro * tanhf(rcs);

    Cs[idx] = cs;  H[idx] = h;  rCs[idx] = rcs;  rH[idx] = rh;
    float cc = c_scalar[0];
    pred[idx] = (h * var1 * cc + rh * var2) / (var1 + var2 * cc);

    GA[(size_t)bc * GATE_N + GC_N + n] = (bf16)h;    // H region of gate input
    RA[(size_t)bc * RIN_K + NND + n] = (bf16)rh;     // rH region of r-gate input
}

// out[b, o, n] = sum_c pred[b,c,n]*conv_w[o,c] + conv_b[o]; flat (64,24,1024)
__global__ void conv_kernel(const float* __restrict__ pred, const float* __restrict__ cw,
                            const float* __restrict__ cb, float* __restrict__ out) {
    int idx = blockIdx.x * blockDim.x + threadIdx.x;  // 64*24*1024
    if (idx >= 64 * 24 * 1024) return;
    int n = idx & 1023;
    int o = (idx >> 10) % 24;
    int b = idx / (24 * 1024);
    float v = pred[((size_t)b * 2 + 0) * NND + n] * cw[o * 2 + 0] +
              pred[((size_t)b * 2 + 1) * NND + n] * cw[o * 2 + 1] + cb[o];
    out[idx] = v;
}

// ------------------------------------------------------------------
extern "C" void kernel_launch(void* const* d_in, const int* in_sizes, int n_in,
                              void* d_out, int out_size, void* d_ws, size_t ws_size,
                              hipStream_t stream) {
    const float* inputs = (const float*)d_in[0];
    const float* adj    = (const float*)d_in[1];
    const float* gc_w   = (const float*)d_in[2];
    const float* Wf  = (const float*)d_in[3];  const float* bfv = (const float*)d_in[4];
    const float* Wi  = (const float*)d_in[5];  const float* biv = (const float*)d_in[6];
    const float* Wo  = (const float*)d_in[7];  const float* bov = (const float*)d_in[8];
    const float* Wc  = (const float*)d_in[9];  const float* bcv = (const float*)d_in[10];
    const float* rWf = (const float*)d_in[11]; const float* rbf = (const float*)d_in[12];
    const float* rWi = (const float*)d_in[13]; const float* rbi = (const float*)d_in[14];
    const float* rWo = (const float*)d_in[15]; const float* rbo = (const float*)d_in[16];
    const float* rWc = (const float*)d_in[17]; const float* rbc = (const float*)d_in[18];
    const float* nw  = (const float*)d_in[19];
    const float* csc = (const float*)d_in[20];
    const float* cw  = (const float*)d_in[21];
    const float* cb  = (const float*)d_in[22];

    char* ws = (char*)d_ws;
    size_t off = 0;
    auto alloc = [&](size_t bytes) { size_t o = off; off += (bytes + 255) & ~(size_t)255; return o; };
    size_t TMP   = alloc(NND * 4);
    size_t ANORM = alloc((size_t)NND * NND * 4);
    size_t A1    = alloc((size_t)NND * NND * 4);
    size_t A2    = alloc((size_t)NND * NND * 4);
    size_t A3    = alloc((size_t)NND * NND * 4);
    size_t EFFW  = alloc((size_t)NND * GC_N * 2);
    size_t WGPK  = alloc((size_t)GATE_N * GATE_N * 2);
    size_t RWGPK = alloc((size_t)RIN_K * GATE_N * 2);
    size_t BG    = alloc(GATE_N * 4);
    size_t RBG   = alloc(GATE_N * 4);
    size_t NBR   = alloc(NND * 4);
    size_t XA    = alloc((size_t)BCR * NND * 2);
    size_t GA    = alloc((size_t)BCR * GATE_N * 2);
    size_t RA    = alloc((size_t)BCR * RIN_K * 2);
    size_t GC    = alloc((size_t)BCR * GC_N * 4);
    size_t G     = alloc((size_t)BCR * GATE_N * 4);
    size_t RG    = alloc((size_t)BCR * GATE_N * 4);
    size_t ST    = alloc((size_t)5 * BCR * NND * 4);  // H, Cs, rH, rCs, pred contiguous
    size_t VAR   = alloc(8 * 4);

    float* pTMP   = (float*)(ws + TMP);
    float* pANORM = (float*)(ws + ANORM);
    float* pA1 = (float*)(ws + A1);
    float* pA2 = (float*)(ws + A2);
    float* pA3 = (float*)(ws + A3);
    bf16* pEFFW  = (bf16*)(ws + EFFW);
    bf16* pWGPK  = (bf16*)(ws + WGPK);
    bf16* pRWGPK = (bf16*)(ws + RWGPK);
    float* pBG  = (float*)(ws + BG);
    float* pRBG = (float*)(ws + RBG);
    float* pNBR = (float*)(ws + NBR);
    bf16* pXA = (bf16*)(ws + XA);
    bf16* pGA = (bf16*)(ws + GA);
    bf16* pRA = (bf16*)(ws + RA);
    float* pGC = (float*)(ws + GC);
    float* pG  = (float*)(ws + G);
    float* pRG = (float*)(ws + RG);
    float* pH   = (float*)(ws + ST);
    float* pCs  = pH + (size_t)BCR * NND;
    float* prH  = pCs + (size_t)BCR * NND;
    float* prCs = prH + (size_t)BCR * NND;
    float* pPred = prCs + (size_t)BCR * NND;
    float* pVAR = (float*)(ws + VAR);

    // ---- one-time precompute ----
    colsum_kernel<<<4, 256, 0, stream>>>(adj, pTMP);
    anorm_kernel<<<4096, 256, 0, stream>>>(adj, pTMP, pANORM);
    clamp1_kernel<<<4096, 256, 0, stream>>>(pANORM, pA1);
    {
        dim3 g(NND / TS, NND / TS), b(TS, TS);
        matmul_clamp_kernel<<<g, b, 0, stream>>>(pA1, pANORM, pA2, NND);
        matmul_clamp_kernel<<<g, b, 0, stream>>>(pA2, pANORM, pA3, NND);
    }
    nbr_kernel<<<4, 256, 0, stream>>>(pA3, nw, pNBR);
    pack_effw_kernel<<<12288, 256, 0, stream>>>(pA1, pA2, pA3, gc_w, pEFFW);
    pack_gate_kernel<<<65536, 256, 0, stream>>>(Wf, Wi, Wo, Wc, pWGPK);
    pack_rgate_kernel<<<32768, 256, 0, stream>>>(rWf, rWi, rWo, rWc, pRWGPK);
    pack_bias_kernel<<<16, 256, 0, stream>>>(bfv, biv, bov, bcv, rbf, rbi, rbo, rbc, pBG, pRBG);

    // zero states (H,Cs,rH,rCs,pred), GA, RA
    zero_u32_kernel<<<2560, 256, 0, stream>>>((uint32_t*)pH, 5 * BCR * NND);
    zero_u32_kernel<<<1024, 256, 0, stream>>>((uint32_t*)pGA, BCR * GATE_N / 2);
    zero_u32_kernel<<<512, 256, 0, stream>>>((uint32_t*)pRA, BCR * RIN_K / 2);

    // ---- recurrence ----
    for (int t = 0; t < LSTEP; ++t) {
        cast_x_kernel<<<512, 256, 0, stream>>>(inputs, t, pXA, pRA);
        // gc = X @ effW^T  -> fp32 GC (ld 3072) + bf16 into GA[:, 0:3072] (ld 4096)
        gemm_bf16_wmma<<<dim3(GC_N / 256, 4), 128, 0, stream>>>(
            pXA, pEFFW, nullptr, pGC, pGA, GATE_N, NND, GC_N);
        zero_u32_kernel<<<1, 32, 0, stream>>>((uint32_t*)pVAR, 8);
        var_acc_kernel<<<dim3(NND / 256, BCR), 256, 0, stream>>>(inputs, NND, LSTEP, t, pVAR);
        var_acc_kernel<<<dim3(GC_N / 256, BCR), 256, 0, stream>>>(pGC, GC_N, 1, 0, pVAR + 4);
        // gates = [gc|H] @ Wg^T + bg
        gemm_bf16_wmma<<<dim3(GATE_N / 256, 4), 128, 0, stream>>>(
            pGA, pWGPK, pBG, pG, (bf16*)nullptr, 0, GATE_N, GATE_N);
        // r-gates = [x|rH] @ rWg^T + rbg
        gemm_bf16_wmma<<<dim3(GATE_N / 256, 4), 128, 0, stream>>>(
            pRA, pRWGPK, pRBG, pRG, (bf16*)nullptr, 0, RIN_K, GATE_N);
        lstm_step_kernel<<<512, 256, 0, stream>>>(pG, pRG, pNBR, pH, pCs, prH, prCs, pPred,
                                                  pGA, pRA, pVAR, csc);
    }

    // ---- output conv ----
    conv_kernel<<<6144, 256, 0, stream>>>(pPred, cw, cb, (float*)d_out);
    (void)in_sizes; (void)n_in; (void)out_size; (void)ws_size;
}
